// GromovWasserstein_18262200943532
// MI455X (gfx1250) — matrix operand order repaired and checked by reading, not compiled, not run on hardware
//
#include <hip/hip_runtime.h>
#include <hip/hip_bf16.h>

// ---------------------------------------------------------------------------
// Gromov-Wasserstein on MI455X (gfx1250), one workgroup per batch element.
// All per-batch state (X, F1, F1^T, F2, F2^T, T1, T2, Xe) lives in the 320KB
// WGP LDS for the whole 15-iteration pipeline; GEMMs use
// v_wmma_f32_16x16x32_bf16 (bf16 A/B, f32 accumulate).
// ---------------------------------------------------------------------------

typedef __bf16 bf16_t;
typedef __attribute__((ext_vector_type(16))) __bf16 v16bf;
typedef __attribute__((ext_vector_type(8)))  __bf16 v8bf;
typedef __attribute__((ext_vector_type(8)))  float  v8f;

#define NN          128
#define LDB         136   // bf16 LDS row stride (pad 8 -> 272B rows, 16B aligned)
#define LDE         132   // f32 Xe row stride (breaks bank conflicts on columns)
#define TAU_F       0.1f
#define GROMOV_ITERS 15
#define SINK_ITERS   15

// A fragment: 16x32 bf16, rows striped within lane (ISA 7.12.2).
// lane<16: row m0+lane, K = k0+{0..7, 16..23}; lane>=16: row m0+lane-16,
// K = k0+{8..15, 24..31}.
__device__ __forceinline__ v16bf load_a_frag(const bf16_t* A, int m0, int k0, int lane) {
  const int lr = lane & 15, half = lane >> 4;
  const bf16_t* p = A + (m0 + lr) * LDB + k0 + half * 8;
  v8bf lo = *(const v8bf*)p;          // K = base .. base+7
  v8bf hi = *(const v8bf*)(p + 16);   // K = base+16 .. base+23
  v16bf r;
#pragma unroll
  for (int i = 0; i < 8; ++i) { r[i] = lo[i]; r[i + 8] = hi[i]; }
  return r;
}

// B fragment: 32x16 bf16; lane holds column n0+(lane&15), K matching A frag
// (lane<16: K=k0+{0..7,16..23}, lane>=16: K=k0+{8..15,24..31}). Source S is stored so
// that S[n*LDB + k] == B[k][n] (column-contiguous reads).
__device__ __forceinline__ v16bf load_b_frag(const bf16_t* S, int n0, int k0, int lane) {
  const int lr = lane & 15, half = lane >> 4;
  const bf16_t* p = S + (n0 + lr) * LDB + k0 + half * 8;
  v8bf lo = *(const v8bf*)p;
  v8bf hi = *(const v8bf*)(p + 16);
  v16bf r;
#pragma unroll
  for (int i = 0; i < 8; ++i) { r[i] = lo[i]; r[i + 8] = hi[i]; }
  return r;
}

__device__ __forceinline__ v8f wmma_bf16(v16bf a, v16bf b, v8f c) {
  return __builtin_amdgcn_wmma_f32_16x16x32_bf16(
      /*neg_a=*/false, a, /*neg_b=*/false, b,
      /*c_mod=*/(short)0, c, /*reuse_a=*/false, /*reuse_b=*/false);
}

__device__ __forceinline__ v8f zero8() {
  v8f z;
#pragma unroll
  for (int i = 0; i < 8; ++i) z[i] = 0.0f;
  return z;
}

__global__ __launch_bounds__(256, 1)
void gw_kernel(const float* __restrict__ Xall, const float* __restrict__ F1all,
               const float* __restrict__ F2all, const float* __restrict__ Kpall,
               float* __restrict__ out, int bs) {
  // ---- LDS layout (~246 KB of the 320 KB WGP LDS) ----
  __shared__ __align__(16) bf16_t sXb [NN * LDB];
  __shared__ __align__(16) bf16_t sF1r[NN * LDB];     // F1 row-major
  __shared__ __align__(16) bf16_t sF1c[NN * LDB];     // F1^T row-major
  __shared__ __align__(16) bf16_t sF2r[NN * LDB];     // F2 row-major
  __shared__ __align__(16) bf16_t sF2c[NN * LDB];     // F2^T row-major
  __shared__ __align__(16) unsigned char sU[2 * NN * LDB * 2]; // T1c+T2c OR Xe
  __shared__ float sRed[256];
  __shared__ float sLu[NN];
  __shared__ float sRv[NN];
  __shared__ float sScale;

  bf16_t* T1c = (bf16_t*)sU;                          // T1 column-major
  bf16_t* T2c = (bf16_t*)(sU + NN * LDB * 2);         // T2 column-major
  float*  Xe  = (float*)sU;                           // reuses T region later

  const int b    = blockIdx.x;
  const int tid  = threadIdx.x;
  const int lane = tid & 31;
  const int wave = tid >> 5;
  const int m0   = wave * 16;             // each wave owns one 16-row band
  const int lr   = lane & 15;
  const int half = lane >> 4;

  const float* Xg  = Xall  + (size_t)b * NN * NN;
  const float* F1g = F1all + (size_t)b * NN * NN;
  const float* F2g = F2all + (size_t)b * NN * NN;
  const float* Kpg = Kpall + (size_t)b * NN * NN;
  float*       og  = out   + (size_t)b * NN * NN;

  // ---- One-time load: f32 -> bf16, plus transposed copies of F1/F2 ----
  for (int idx = tid; idx < NN * NN; idx += 256) {
    const int m = idx >> 7, n = idx & 127;
    const float x  = Xg[idx];
    const float f1 = F1g[idx];
    const float f2 = F2g[idx];
    sXb [m * LDB + n] = (bf16_t)x;
    sF1r[m * LDB + n] = (bf16_t)f1;
    sF1c[n * LDB + m] = (bf16_t)f1;
    sF2r[m * LDB + n] = (bf16_t)f2;
    sF2c[n * LDB + m] = (bf16_t)f2;
  }
  __syncthreads();

  for (int it = 0; it < GROMOV_ITERS; ++it) {
    // Pull the Kp panel toward L2/L0 for the stage-2 accumulator init.
    __builtin_prefetch(Kpg + tid * 32, 0, 1);
    __builtin_prefetch(Kpg + 8192 + tid * 32, 0, 1);

    // ---- Stage 1: T1 = X*F2, T2 = X*F2^T (stored column-major, bf16) ----
    {
      v16bf ax[4];
#pragma unroll
      for (int kk = 0; kk < 4; ++kk) ax[kk] = load_a_frag(sXb, m0, kk * 32, lane);
#pragma unroll
      for (int nt = 0; nt < 8; ++nt) {
        const int n0 = nt * 16;
        v8f acc1 = zero8();
        v8f acc2 = zero8();
#pragma unroll
        for (int kk = 0; kk < 4; ++kk) {
          // B = F2:   B[k][n] = F2[k][n]  -> contiguous in sF2c
          acc1 = wmma_bf16(ax[kk], load_b_frag(sF2c, n0, kk * 32, lane), acc1);
          // B = F2^T: B[k][n] = F2[n][k]  -> contiguous in sF2r
          acc2 = wmma_bf16(ax[kk], load_b_frag(sF2r, n0, kk * 32, lane), acc2);
        }
#pragma unroll
        for (int v = 0; v < 8; ++v) {
          const int m = m0 + half * 8 + v;       // C/D layout: row = vgpr + 8*half
          T1c[(n0 + lr) * LDB + m] = (bf16_t)acc1[v];
          T2c[(n0 + lr) * LDB + m] = (bf16_t)acc2[v];
        }
      }
    }
    __syncthreads();

    // ---- Stage 2: C = Kp + F1*T1 + F1^T*T2 ; C kept in registers ----
    v8f accs[8];
    float lmax = 0.0f;
#pragma unroll
    for (int nt = 0; nt < 8; ++nt) {
      const int n0 = nt * 16;
      v8f acc;
#pragma unroll
      for (int v = 0; v < 8; ++v)
        acc[v] = Kpg[(m0 + half * 8 + v) * NN + n0 + lr];
#pragma unroll
      for (int kk = 0; kk < 4; ++kk) {
        acc = wmma_bf16(load_a_frag(sF1r, m0, kk * 32, lane),
                        load_b_frag(T1c,  n0, kk * 32, lane), acc);
        acc = wmma_bf16(load_a_frag(sF1c, m0, kk * 32, lane),
                        load_b_frag(T2c,  n0, kk * 32, lane), acc);
      }
      accs[nt] = acc;
#pragma unroll
      for (int v = 0; v < 8; ++v) lmax = fmaxf(lmax, fabsf(acc[v]));
    }

    // ---- L-inf norm over the 128x128 block ----
    sRed[tid] = lmax;
    __syncthreads();            // also: all T1c/T2c consumers are done here
    if (tid == 0) {
      float mx = 0.0f;
      for (int i = 0; i < 256; ++i) mx = fmaxf(mx, sRed[i]);
      sScale = 10.0f / mx;      // exp(-(-C/mx)/TAU) = exp(C/(mx*TAU)), TAU=0.1
    }
    __syncthreads();
    const float scl = sScale;

    // ---- Xe = exp(C*scl), written into reused LDS region (f32) ----
#pragma unroll
    for (int nt = 0; nt < 8; ++nt) {
#pragma unroll
      for (int v = 0; v < 8; ++v) {
        const int m = m0 + half * 8 + v;
        const int n = nt * 16 + lr;
        Xe[m * LDE + n] = __expf(accs[nt][v] * scl);
      }
    }
    if (tid < NN) sRv[tid] = 1.0f;
    __syncthreads();

    // ---- Partial Sinkhorn (15 matvec pairs on VALU) ----
    for (int s = 0; s < SINK_ITERS; ++s) {
      if (tid < NN) {
        float acc = 0.0f;
#pragma unroll 8
        for (int n = 0; n < NN; ++n) acc += Xe[tid * LDE + n] * sRv[n];
        sLu[tid] = 1.0f / acc;
      }
      __syncthreads();
      if (tid < NN) {
        float acc = 0.0f;
#pragma unroll 8
        for (int m = 0; m < NN; ++m) acc += Xe[m * LDE + tid] * sLu[m];
        const float r = 1.0f / acc;
        sRv[tid] = fminf(r, 1.0f);
      }
      __syncthreads();
    }

    // ---- X = diag(lu) * Xe * diag(rv) ----
    if (it + 1 < GROMOV_ITERS) {
      for (int idx = tid; idx < NN * NN; idx += 256) {
        const int m = idx >> 7, n = idx & 127;
        sXb[m * LDB + n] = (bf16_t)(sLu[m] * Xe[m * LDE + n] * sRv[n]);
      }
      __syncthreads();
    } else {
      for (int idx = tid; idx < NN * NN; idx += 256) {
        const int m = idx >> 7, n = idx & 127;
        og[idx] = sLu[m] * Xe[m * LDE + n] * sRv[n];
      }
    }
  }
}

extern "C" void kernel_launch(void* const* d_in, const int* in_sizes, int n_in,
                              void* d_out, int out_size, void* d_ws, size_t ws_size,
                              hipStream_t stream) {
  const float* X  = (const float*)d_in[0];
  const float* F1 = (const float*)d_in[1];
  const float* F2 = (const float*)d_in[2];
  const float* Kp = (const float*)d_in[3];
  float* out = (float*)d_out;
  const int bs = in_sizes[0] / (NN * NN);   // 2048
  gw_kernel<<<dim3(bs), dim3(256), 0, stream>>>(X, F1, F2, Kp, out, bs);
}
